// GNN_node_36369783063008
// MI455X (gfx1250) — matrix operand order repaired and checked by reading, not compiled, not run on hardware
//
#include <hip/hip_runtime.h>

typedef __attribute__((ext_vector_type(16))) __bf16 v16bf;
typedef __attribute__((ext_vector_type(8)))  __bf16 v8bf;
typedef __attribute__((ext_vector_type(8)))  float  v8f;

#define DFEAT 128
#define NLAYER 5

// ---------------- utility kernels ----------------

__global__ __launch_bounds__(256) void k_zero_f32(float* __restrict__ p, long long n4) {
    long long i = (long long)blockIdx.x * blockDim.x + threadIdx.x;
    if (i < n4) ((float4*)p)[i] = make_float4(0.f, 0.f, 0.f, 0.f);
}

__global__ __launch_bounds__(256) void k_deg_count(const long long* __restrict__ ei,
                                                   float* __restrict__ deg, int E) {
    int e = blockIdx.x * blockDim.x + threadIdx.x;
    if (e < E) atomicAdd(&deg[ei[e]], 1.0f);
}

__global__ __launch_bounds__(256) void k_deg_fin(float* __restrict__ deg,
                                                 float* __restrict__ dinv, int N) {
    int n = blockIdx.x * blockDim.x + threadIdx.x;
    if (n < N) { float d = deg[n] + 1.0f; deg[n] = d; dinv[n] = rsqrtf(d); }
}

__global__ __launch_bounds__(256) void k_edge_norm(const long long* __restrict__ ei,
                                                   const float* __restrict__ dinv,
                                                   float* __restrict__ nrm, int E) {
    int e = blockIdx.x * blockDim.x + threadIdx.x;
    if (e < E) nrm[e] = dinv[ei[e]] * dinv[ei[E + e]];
}

__global__ __launch_bounds__(256) void k_f32_to_bf16(const float* __restrict__ in,
                                                     __bf16* __restrict__ out, long long n) {
    long long i = (long long)blockIdx.x * blockDim.x + threadIdx.x;
    if (i < n) out[i] = (__bf16)in[i];
}

// edge_attr int32 (values 0..3) -> uint8, cuts HBM streaming 4x
__global__ __launch_bounds__(256) void k_ea_to_u8(const int* __restrict__ ea,
                                                  unsigned char* __restrict__ out, long long n4) {
    long long i = (long long)blockIdx.x * blockDim.x + threadIdx.x;
    if (i < n4) {
        int4 v = ((const int4*)ea)[i];
        ((uchar4*)out)[i] = make_uchar4((unsigned char)v.x, (unsigned char)v.y,
                                        (unsigned char)v.z, (unsigned char)v.w);
    }
}

// ---------------- GEMM: hl = h @ W^T + b  (bf16 WMMA, f32 accum) ----------------
// One wave computes one 16x16 output tile. Block = 8 waves = one 16-row stripe
// across all 8 column tiles (A reuse via L0).
__global__ __launch_bounds__(256) void k_gemm_bf16(const __bf16* __restrict__ A,   // [N,128] bf16
                                                   const __bf16* __restrict__ W,   // [128,128] bf16 row-major
                                                   const float* __restrict__ bias, // [128]
                                                   float* __restrict__ out,        // [N,128] f32
                                                   int N) {
    const int lane = threadIdx.x & 31;
    const int wave = threadIdx.x >> 5;
    const int tile = blockIdx.x * 8 + wave;
    const int tr = tile >> 3;      // row-tile index
    const int tc = tile & 7;       // col-tile index (dout/16)
    if (tr * 16 >= N) return;
    const int half = lane >> 4;    // 0 or 1
    const int r    = lane & 15;

    // A fragment: row = tr*16 + r; elems 0..7 = K kb+half*8.., elems 8..15 = K kb+16+half*8..
    const __bf16* Ab = A + (size_t)(tr * 16 + r) * DFEAT + half * 8;
    // B fragment: col n = r -> W row (tc*16 + r); elems 0..7 = K kb+half*8.., elems 8..15 = K kb+16+half*8..
    const __bf16* Bb = W + (size_t)(tc * 16 + r) * DFEAT + half * 8;

    v8f acc = {};
#pragma unroll
    for (int kb = 0; kb < DFEAT; kb += 32) {
        v8bf alo = *(const v8bf*)(Ab + kb);
        v8bf ahi = *(const v8bf*)(Ab + kb + 16);
        v16bf a, b;
        v8bf blo = *(const v8bf*)(Bb + kb);
        v8bf bhi = *(const v8bf*)(Bb + kb + 16);
#pragma unroll
        for (int i = 0; i < 8; ++i) { a[i] = alo[i]; a[i + 8] = ahi[i]; b[i] = blo[i]; b[i + 8] = bhi[i]; }
        acc = __builtin_amdgcn_wmma_f32_16x16x32_bf16(false, a, false, b,
                                                      (short)0, acc, false, false);
    }
    // D layout: VGPR v -> row tr*16 + v + 8*half, col tc*16 + r
    const float bv  = bias[tc * 16 + r];
    const int   col = tc * 16 + r;
    const int  row0 = tr * 16 + half * 8;
#pragma unroll
    for (int v = 0; v < 8; ++v)
        out[(size_t)(row0 + v) * DFEAT + col] = acc[v] + bv;
}

// ---------------- edge phase: msg = norm * relu(hl[row] + e); agg[col] += msg ----------------
// One thread handles one edge x 4 features (float4 gather, uchar4 attr, 4 f32 atomics).
__global__ __launch_bounds__(256) void k_edge(const float* __restrict__ hl,
                                              const long long* __restrict__ ei,
                                              const unsigned char* __restrict__ ea8,
                                              const float* __restrict__ nrm,
                                              float* __restrict__ agg, int E) {
    long long t = (long long)blockIdx.x * blockDim.x + threadIdx.x;
    if (t >= (long long)E * 32) return;
    int e  = (int)(t >> 5);
    int dq = ((int)t & 31) * 4;
    long long row = ei[e];
    long long col = ei[E + e];
    float s = nrm[e];
    float4 hv = *(const float4*)(hl + row * DFEAT + dq);
    unsigned int a = *(const unsigned int*)(ea8 + (size_t)e * DFEAT + dq);
    float m0 = fmaxf(hv.x + (float)( a        & 0xFF), 0.f) * s;
    float m1 = fmaxf(hv.y + (float)((a >> 8)  & 0xFF), 0.f) * s;
    float m2 = fmaxf(hv.z + (float)((a >> 16) & 0xFF), 0.f) * s;
    float m3 = fmaxf(hv.w + (float)( a >> 24        ), 0.f) * s;
    float* ap = agg + col * DFEAT + dq;
    atomicAdd(ap + 0, m0);
    atomicAdd(ap + 1, m1);
    atomicAdd(ap + 2, m2);
    atomicAdd(ap + 3, m3);
}

// ---------------- node update + BatchNorm (eval) + optional relu ----------------
__global__ __launch_bounds__(256) void k_node(const float* __restrict__ agg,
                                              const float* __restrict__ hl,
                                              const float* __restrict__ deg,
                                              const float* __restrict__ root,
                                              const float* __restrict__ gamma,
                                              const float* __restrict__ beta,
                                              const float* __restrict__ mean,
                                              const float* __restrict__ var,
                                              float* __restrict__ out,
                                              int N, int do_relu) {
    int t = blockIdx.x * blockDim.x + threadIdx.x;
    if (t >= N * DFEAT) return;
    int d = t & (DFEAT - 1);
    float dg = deg[t >> 7];
    float h = agg[t] + fmaxf(hl[t] + root[d], 0.f) / dg;
    h = gamma[d] * (h - mean[d]) * rsqrtf(var[d] + 1e-5f) + beta[d];
    if (do_relu) h = fmaxf(h, 0.f);
    out[t] = h;
}

// ---------------- host launcher ----------------
extern "C" void kernel_launch(void* const* d_in, const int* in_sizes, int n_in,
                              void* d_out, int out_size, void* d_ws, size_t ws_size,
                              hipStream_t stream) {
    const int D = DFEAT;
    const int N = in_sizes[0] / D;   // x: [N,128] f32
    const int E = in_sizes[2] / D;   // edge_attr: [E,128] i32

    const float*     x     = (const float*)d_in[0];
    const long long* ei    = (const long long*)d_in[1];  // [2,E] int64
    const int*       ea    = (const int*)d_in[2];
    const float*     Wf    = (const float*)d_in[3];      // [L,128,128]
    const float*     bvec  = (const float*)d_in[4];      // [L,128]
    const float*     rootf = (const float*)d_in[5];
    const float*     gam   = (const float*)d_in[6];
    const float*     bet   = (const float*)d_in[7];
    const float*     mu    = (const float*)d_in[8];
    const float*     va    = (const float*)d_in[9];
    float*           out   = (float*)d_out;

    char* wp = (char*)d_ws;
    auto take = [&](size_t bytes) { char* p = wp; wp += (bytes + 255) & ~(size_t)255; return p; };
    float*         deg  = (float*)take((size_t)N * 4);
    float*         dinv = (float*)take((size_t)N * 4);
    float*         nrm  = (float*)take((size_t)E * 4);
    __bf16*        Wb   = (__bf16*)take((size_t)NLAYER * D * D * 2);
    __bf16*        hb   = (__bf16*)take((size_t)N * D * 2);
    unsigned char* ea8  = (unsigned char*)take((size_t)E * D);
    float*         hl   = (float*)take((size_t)N * D * 4);
    float*         agg  = (float*)take((size_t)N * D * 4);
    float*         hA   = (float*)take((size_t)N * D * 4);
    float*         hB   = (float*)take((size_t)N * D * 4);

    const int TB = 256;
    const long long nd = (long long)N * D;

    // graph-invariant precompute (runs every call; deterministic)
    k_zero_f32<<<(int)((N / 4 + TB - 1) / TB), TB, 0, stream>>>(deg, N / 4);
    k_deg_count<<<(E + TB - 1) / TB, TB, 0, stream>>>(ei, deg, E);
    k_deg_fin<<<(N + TB - 1) / TB, TB, 0, stream>>>(deg, dinv, N);
    k_edge_norm<<<(E + TB - 1) / TB, TB, 0, stream>>>(ei, dinv, nrm, E);

    long long wcnt = (long long)NLAYER * D * D;
    k_f32_to_bf16<<<(int)((wcnt + TB - 1) / TB), TB, 0, stream>>>(Wf, Wb, wcnt);
    long long ea4 = (long long)E * D / 4;
    k_ea_to_u8<<<(int)((ea4 + TB - 1) / TB), TB, 0, stream>>>(ea, ea8, ea4);

    const float* hin = x;
    for (int l = 0; l < NLAYER; ++l) {
        float* hout = (l == NLAYER - 1) ? out : ((l & 1) ? hB : hA);

        k_f32_to_bf16<<<(int)((nd + TB - 1) / TB), TB, 0, stream>>>(hin, hb, nd);

        int tiles = ((N + 15) / 16) * 8;
        k_gemm_bf16<<<(tiles + 7) / 8, TB, 0, stream>>>(hb, Wb + (size_t)l * D * D,
                                                        bvec + l * D, hl, N);

        k_zero_f32<<<(int)((nd / 4 + TB - 1) / TB), TB, 0, stream>>>(agg, nd / 4);

        long long et = (long long)E * 32;
        k_edge<<<(int)((et + TB - 1) / TB), TB, 0, stream>>>(hl, ei, ea8, nrm, agg, E);

        k_node<<<(int)((nd + TB - 1) / TB), TB, 0, stream>>>(agg, hl, deg, rootf + l * D,
                                                             gam + l * D, bet + l * D,
                                                             mu + l * D, va + l * D,
                                                             hout, N, l < NLAYER - 1);
        hin = hout;
    }
}
